// NonLocalBlock_8108898255460
// MI455X (gfx1250) — matrix operand order repaired and checked by reading, not compiled, not run on hardware
//
#include <hip/hip_runtime.h>

typedef __attribute__((ext_vector_type(16))) _Float16 v16h;
typedef __attribute__((ext_vector_type(8)))  float    v8f;

#define BATCH 4
#define CIN   64
#define C2    32
#define HH    128
#define WW    128
#define HW    (HH*WW)        // 16384 queries per batch
#define HW4   (HW/4)         // 4096 pooled keys per batch
#define TILES_PER_B (HW/16)  // 1024 query tiles of 16

#define LOG2E 1.44269504088896340736f

// ---------------------------------------------------------------------------
// Transposed flash attention, all operands computed/packed so the hot loop has
// NO LDS and (almost) no cross-lane traffic:
//   S^T = phi^T(16m x 32c) . theta^T(32c x 16n)   [theta pre-scaled by log2e]
//   y^T = g^T (16c x 32m) . P^T    (32m x 16n)
// Fragments (CDNA5 ISA 7.12.2, wave32, f16):
//   A 16x32: lane = row + 16*((K>>4)&1); half = K & 15 (contiguous K, not the ISA table)
//   B 32x16: lane = col + 16*((K>>4)&1); half = K & 15 (same contiguous order as A; not the ISA table)
// Key permutation inside each 32-key chunk: key i lives at phi-fragment
// j=(i>>3)&1, row r=(i&7)+8*((i>>4)&1).  With that placement the C-layout of
// S^T coincides lane-for-lane with the B-layout of P^T, so P^T is built with
// plain per-lane f32->f16 converts.  g uses K=i directly (consistent).
// ---------------------------------------------------------------------------

__device__ __forceinline__ float permx16(float v) {   // lane <-> lane^16
  int i = __builtin_bit_cast(int, v);
  int r = __builtin_amdgcn_permlanex16(i, i, 0x76543210, 0xFEDCBA98, false, false);
  return __builtin_bit_cast(float, r);
}

// Kernel A1: theta projection (pre-scaled by log2e), B-frags (K=ch, col=query)
__global__ __launch_bounds__(256) void proj_theta_kernel(
    const float* __restrict__ x, const float* __restrict__ w,
    const float* __restrict__ bias, _Float16* __restrict__ theta_pk)
{
  __shared__ float ws[C2*CIN];
  __shared__ float bs[C2];
  for (int i = threadIdx.x; i < C2*CIN; i += 256) ws[i] = w[i];
  if (threadIdx.x < C2) bs[threadIdx.x] = bias[threadIdx.x];
  __syncthreads();

  int gid = blockIdx.x * 256 + threadIdx.x;      // over B*HW
  int b = gid / HW, p = gid % HW;
  const float* xp = x + (size_t)b * CIN * HW + p;
  float xr[CIN];
  #pragma unroll
  for (int c = 0; c < CIN; ++c) xr[c] = xp[(size_t)c * HW];

  float acc[C2];
  #pragma unroll
  for (int o = 0; o < C2; ++o) {
    float a = bs[o];
    #pragma unroll
    for (int c = 0; c < CIN; ++c) a = fmaf(xr[c], ws[o*CIN + c], a);
    acc[o] = a * LOG2E;                          // scores land in exp2 domain
  }
  // B-frag: lane (p&15) holds K=c 0..15 (h=c); lane (p&15)+16 holds 16..31
  v16h lo, hi;
  #pragma unroll
  for (int h = 0; h < 16; ++h) {
    lo[h] = (_Float16)acc[h];
    hi[h] = (_Float16)acc[h + 16];
  }
  _Float16* base = theta_pk + ((size_t)b * HW + (size_t)(p & ~15)) * 32;
  *(v16h*)(base + (p & 15) * 16)        = lo;
  *(v16h*)(base + ((p & 15) + 16) * 16) = hi;
}

// Kernel A2: phi & g projections + 2x2 maxpool.
// phi: A-frags with the key permutation described above  [B][HW4/16][512]
// g:   A-frags for g^T (row=channel&15, K=key&31)        [B][HW4/32][2][512]
__global__ __launch_bounds__(256) void proj_phig_kernel(
    const float* __restrict__ x,
    const float* __restrict__ w_phi, const float* __restrict__ b_phi,
    const float* __restrict__ w_g,   const float* __restrict__ b_g,
    _Float16* __restrict__ phi_pk, _Float16* __restrict__ g_pk)
{
  __shared__ float wsp[C2*CIN];
  __shared__ float wsg[C2*CIN];
  __shared__ float bsp[C2], bsg[C2];
  for (int i = threadIdx.x; i < C2*CIN; i += 256) { wsp[i] = w_phi[i]; wsg[i] = w_g[i]; }
  if (threadIdx.x < C2) { bsp[threadIdx.x] = b_phi[threadIdx.x]; bsg[threadIdx.x] = b_g[threadIdx.x]; }
  __syncthreads();

  int gid = blockIdx.x * 256 + threadIdx.x;      // over B*HW4
  int b = gid / HW4, m = gid % HW4;
  int ph = m / (WW/2), pw = m % (WW/2);

  float mphi[C2], mg[C2];
  #pragma unroll
  for (int o = 0; o < C2; ++o) { mphi[o] = -1e30f; mg[o] = -1e30f; }

  for (int d = 0; d < 4; ++d) {                  // 2x2 pool window
    int p = (ph*2 + (d >> 1)) * WW + pw*2 + (d & 1);
    const float* xp = x + (size_t)b * CIN * HW + p;
    float xr[CIN];
    #pragma unroll
    for (int c = 0; c < CIN; ++c) xr[c] = xp[(size_t)c * HW];
    #pragma unroll
    for (int o = 0; o < C2; ++o) {
      float ap = 0.f, ag = 0.f;
      #pragma unroll
      for (int c = 0; c < CIN; ++c) {
        ap = fmaf(xr[c], wsp[o*CIN + c], ap);
        ag = fmaf(xr[c], wsg[o*CIN + c], ag);
      }
      mphi[o] = fmaxf(mphi[o], ap);
      mg[o]   = fmaxf(mg[o],   ag);
    }
  }
  // phi A-frag with key permutation: i = m&31 ->
  //   frag = (m>>5)*2 + ((m>>3)&1), row r = (m&7) + 8*((m>>4)&1)
  {
    v16h lo, hi;
    #pragma unroll
    for (int h = 0; h < 16; ++h) {
      int k = h;
      lo[h] = (_Float16)(mphi[k]      + bsp[k]);      // lane r   : c 0-15
      hi[h] = (_Float16)(mphi[k + 16] + bsp[k + 16]); // lane r+16: c 16-31
    }
    int frag = ((m >> 5) << 1) + ((m >> 3) & 1);
    int r    = (m & 7) + 8 * ((m >> 4) & 1);
    _Float16* base = phi_pk + (size_t)b * (C2*HW4) + (size_t)frag * 512;
    *(v16h*)(base + r * 16)        = lo;
    *(v16h*)(base + (r + 16) * 16) = hi;
  }
  // g^T A-frag: row = c&15, K = m&31 -> lane = (c&15)+16*((m>>4)&1),
  // half = m&15, frag = [m>>5][c>>4]
  {
    _Float16* base = g_pk + (size_t)b * (C2*HW4) + (size_t)(m >> 5) * 1024;
    int lsel = (m >> 4) & 1;
    int h = m & 15;
    #pragma unroll
    for (int c = 0; c < C2; ++c) {
      base[(size_t)(c >> 4) * 512 + ((c & 15) + 16*lsel) * 16 + h] =
          (_Float16)(mg[c] + bsg[c]);
    }
  }
}

// Kernel B: fused transposed flash attention. 1 wave = one 16-query tile.
// Hot loop: 8 b128 loads, 4 WMMAs, per-lane softmax, 1 permlanex16. No LDS.
__global__ __launch_bounds__(256) void attn_kernel(
    const _Float16* __restrict__ theta_pk,
    const _Float16* __restrict__ phi_pk,
    const _Float16* __restrict__ g_pk,
    float* __restrict__ y_ws)                       // y^T: [B][C2][HW]
{
  int lane = threadIdx.x & 31;
  int wave = threadIdx.x >> 5;
  int tile_g = blockIdx.x * 8 + wave;               // 0 .. B*1024-1
  int b = tile_g >> 10;
  int tile_b = tile_g & 1023;
  int hihalf = lane >> 4;

  v16h thB = *(const v16h*)(theta_pk + (size_t)b * HW * 32
                            + (size_t)tile_b * 512 + lane * 16);
  const _Float16* phibase = phi_pk + (size_t)b * (C2*HW4);
  const _Float16* gbase   = g_pk   + (size_t)b * (C2*HW4);

  float row_m = -1e30f;       // per-lane (query = lane&15, max over keys of both halves)
  float row_l = 0.f;          // per-lane partial softmax denom
  v8f yacc0 = {}, yacc1 = {};
  v8f zero = {};

  // register double-buffer for streamed phi/g fragments
  v16h pa0 = *(const v16h*)(phibase + 0*512 + lane * 16);
  v16h pa1 = *(const v16h*)(phibase + 1*512 + lane * 16);
  v16h ga0 = *(const v16h*)(gbase   + 0     + lane * 16);
  v16h ga1 = *(const v16h*)(gbase   + 512   + lane * 16);

  for (int mc = 0; mc < HW4/32; ++mc) {
    v16h cpa0 = pa0, cpa1 = pa1, cga0 = ga0, cga1 = ga1;
    int mcn = (mc + 1) & (HW4/32 - 1);              // wrap: always in-bounds
    pa0 = *(const v16h*)(phibase + (size_t)(2*mcn + 0) * 512 + lane * 16);
    pa1 = *(const v16h*)(phibase + (size_t)(2*mcn + 1) * 512 + lane * 16);
    ga0 = *(const v16h*)(gbase + (size_t)mcn * 1024 +       lane * 16);
    ga1 = *(const v16h*)(gbase + (size_t)mcn * 1024 + 512 + lane * 16);

    // S^T (exp2 domain): rows = keys on vgprs, cols = queries on lanes
    v8f s0 = __builtin_amdgcn_wmma_f32_16x16x32_f16(false, cpa0, false, thB,
                                                    (short)0, zero, false, false);
    v8f s1 = __builtin_amdgcn_wmma_f32_16x16x32_f16(false, cpa1, false, thB,
                                                    (short)0, zero, false, false);
    // per-lane max tree over this lane's 16 keys, then cross-half combine
    float t[8];
    #pragma unroll
    for (int v = 0; v < 8; ++v) t[v] = fmaxf(s0[v], s1[v]);
    float m01 = fmaxf(t[0], t[1]), m23 = fmaxf(t[2], t[3]);
    float m45 = fmaxf(t[4], t[5]), m67 = fmaxf(t[6], t[7]);
    float mx = fmaxf(fmaxf(m01, m23), fmaxf(m45, m67));
    mx = fmaxf(mx, permx16(mx));                    // v_permlanex16 (VALU)
    float nm    = fmaxf(row_m, mx);
    float alpha = __builtin_amdgcn_exp2f(row_m - nm);
    row_m = nm;

    float e0[8], e1[8];
    #pragma unroll
    for (int v = 0; v < 8; ++v) {
      e0[v] = __builtin_amdgcn_exp2f(s0[v] - nm);
      e1[v] = __builtin_amdgcn_exp2f(s1[v] - nm);
    }
    float a0 = (e0[0]+e0[1]) + (e0[2]+e0[3]);
    float a1 = (e0[4]+e0[5]) + (e0[6]+e0[7]);
    float a2 = (e1[0]+e1[1]) + (e1[2]+e1[3]);
    float a3 = (e1[4]+e1[5]) + (e1[6]+e1[7]);
    row_l = row_l * alpha + ((a0+a1) + (a2+a3));    // partial; combined at end
    #pragma unroll
    for (int v = 0; v < 8; ++v) { yacc0[v] *= alpha; yacc1[v] *= alpha; }

    // P^T B-frag: with the key permutation this is a pure per-lane repack
    v16h pB;
    #pragma unroll
    for (int v = 0; v < 8; ++v) {
      pB[v]     = (_Float16)e0[v];                  // K = (lane>=16)*16 + v
      pB[v + 8] = (_Float16)e1[v];                  // K = (lane>=16)*16 + 8+v
    }
    // y^T += g^T . P^T  (rows = channels on vgprs, cols = queries on lanes)
    yacc0 = __builtin_amdgcn_wmma_f32_16x16x32_f16(false, cga0, false, pB,
                                                   (short)0, yacc0, false, false);
    yacc1 = __builtin_amdgcn_wmma_f32_16x16x32_f16(false, cga1, false, pB,
                                                   (short)0, yacc1, false, false);
  }
  row_l += permx16(row_l);                          // total softmax denom
  float inv = 1.0f / row_l;
  int q = tile_b * 16 + (lane & 15);
  #pragma unroll
  for (int v = 0; v < 8; ++v) {
    int c0 = v + 8*hihalf;
    y_ws[((size_t)(b*C2) + c0)      * HW + q] = yacc0[v] * inv;
    y_ws[((size_t)(b*C2) + c0 + 16) * HW + q] = yacc1[v] * inv;
  }
}

// Kernel C: out = W_out * y + b_out + x   (y^T is channel-major: coalesced)
__global__ __launch_bounds__(256) void out_kernel(
    const float* __restrict__ y_ws, const float* __restrict__ w_out,
    const float* __restrict__ b_out, const float* __restrict__ x,
    float* __restrict__ out)
{
  __shared__ float ws[CIN*C2];
  __shared__ float bs[CIN];
  for (int i = threadIdx.x; i < CIN*C2; i += 256) ws[i] = w_out[i];
  for (int i = threadIdx.x; i < CIN;    i += 256) bs[i] = b_out[i];
  __syncthreads();

  int gid = blockIdx.x * 256 + threadIdx.x;
  int b = gid / HW, p = gid % HW;
  float yr[C2];
  #pragma unroll
  for (int c = 0; c < C2; ++c) yr[c] = y_ws[((size_t)(b*C2) + c) * HW + p];
  const float* xp = x   + (size_t)b * CIN * HW + p;
  float*       op = out + (size_t)b * CIN * HW + p;
  #pragma unroll
  for (int o = 0; o < CIN; ++o) {
    float a = bs[o];
    #pragma unroll
    for (int c = 0; c < C2; ++c) a = fmaf(yr[c], ws[o*C2 + c], a);
    op[(size_t)o * HW] = a + xp[(size_t)o * HW];
  }
}

extern "C" void kernel_launch(void* const* d_in, const int* in_sizes, int n_in,
                              void* d_out, int out_size, void* d_ws, size_t ws_size,
                              hipStream_t stream) {
  const float* x       = (const float*)d_in[0];
  const float* w_theta = (const float*)d_in[1];
  const float* b_theta = (const float*)d_in[2];
  const float* w_phi   = (const float*)d_in[3];
  const float* b_phi   = (const float*)d_in[4];
  const float* w_g     = (const float*)d_in[5];
  const float* b_g     = (const float*)d_in[6];
  const float* w_out   = (const float*)d_in[7];
  const float* b_out   = (const float*)d_in[8];
  float* out = (float*)d_out;

  char* ws = (char*)d_ws;
  _Float16* theta_pk = (_Float16*)(ws);                 // 4 MB
  _Float16* phi_pk   = (_Float16*)(ws + (4u << 20));    // 1 MB
  _Float16* g_pk     = (_Float16*)(ws + (5u << 20));    // 1 MB
  float*    y_ws     = (float*)   (ws + (6u << 20));    // 8 MB

  proj_theta_kernel<<<BATCH*HW/256, 256, 0, stream>>>(x, w_theta, b_theta, theta_pk);
  proj_phig_kernel<<<BATCH*HW4/256, 256, 0, stream>>>(x, w_phi, b_phi, w_g, b_g,
                                                      phi_pk, g_pk);
  attn_kernel<<<BATCH*TILES_PER_B/8, 256, 0, stream>>>(theta_pk, phi_pk, g_pk, y_ws);
  out_kernel<<<BATCH*HW/256, 256, 0, stream>>>(y_ws, w_out, b_out, x, out);
}
